// FocalModulation_52810917872298
// MI455X (gfx1250) — matrix operand order repaired and checked by reading, not compiled, not run on hardware
//
#include <hip/hip_runtime.h>
#include <hip/hip_bf16.h>
#include <math.h>

#define DIMD 1024
#define SEQ  4096
#define BATCH 8

typedef __attribute__((ext_vector_type(16))) __bf16          bf16x16;
typedef __attribute__((ext_vector_type(2)))  __bf16          bf16x2;
typedef __attribute__((ext_vector_type(2)))  float           f32x2;
typedef __attribute__((ext_vector_type(8)))  float           v8f;
typedef __attribute__((ext_vector_type(8)))  unsigned short  u16x8;

union Frag { u16x8 h[2]; bf16x16 v; };

// native bf16 conversion helpers
__device__ __forceinline__ unsigned short f2bf(float f) {
  __bf16 h = (__bf16)f;
  return __builtin_bit_cast(unsigned short, h);
}
// packed pair -> exactly one v_cvt_pk_bf16_f32, no repacking ALU
__device__ __forceinline__ unsigned int pack2bf(float a, float b) {
  f32x2 f; f.x = a; f.y = b;
  bf16x2 h = __builtin_convertvector(f, bf16x2);
  return __builtin_bit_cast(unsigned int, h);
}
__device__ __forceinline__ float bf2f(unsigned short h) {
  return __uint_as_float(((unsigned int)h) << 16);
}
__device__ __forceinline__ float gelu(float x) {
  return 0.5f * x * (1.0f + erff(x * 0.70710678118654752f));
}

// ---------------------------------------------------------------------------
// bf16 WMMA GEMM: C(M,N) = A(M,K) @ B(K,N) (+bias), B pre-staged as (N,K) bf16.
// Tile 128x128x32, 256 threads (8 wave32), each wave = 4x2 grid of 16x16 tiles.
// Ping-pong LDS buffers -> single barrier per K-step.
// EPI 0: f32 store + bias (final output)
// EPI 1: split: col<1024 -> query bf16 (BS,D); col>=1024 -> ctx bf16 (B,D,S)
// EPI 2: (acc+bias)*query -> y bf16 (BS,D)
// ---------------------------------------------------------------------------
constexpr int BM = 128, BN = 128, BK = 32, LDT = BK + 8;   // LDT=40 -> 80B rows

template<bool ABF16, int EPI>
__global__ __launch_bounds__(256) void gemm_wmma(
    const void* __restrict__ Av,
    const unsigned short* __restrict__ Bw,
    const float* __restrict__ bias,
    const unsigned short* __restrict__ qbuf,
    void* __restrict__ out0, void* __restrict__ out1,
    int M, int N, int K)
{
  __shared__ unsigned short As[2][BM * LDT];
  __shared__ unsigned short Bs[2][BN * LDT];
  const int tid  = threadIdx.x;
  const int lane = tid & 31, wave = tid >> 5;
  const int wm = wave & 1, wn = wave >> 1;          // 2 x 4 wave grid
  const int m0 = blockIdx.y * BM, n0 = blockIdx.x * BN;

  v8f acc[4][2];
#pragma unroll
  for (int i = 0; i < 4; i++)
#pragma unroll
    for (int j = 0; j < 2; j++)
#pragma unroll
      for (int v = 0; v < 8; v++) acc[i][j][v] = 0.0f;

  float4 aregf[4];
  u16x8  aregh[2];
  u16x8  bregh[2];

  const int ar8 = tid >> 3, ac4 = (tid & 7) << 2;   // f32 A: 4 float4 per thread
  const int hr  = tid >> 2, hc8 = (tid & 3) << 3;   // bf16 A/B: 2 u16x8 per thread

  auto fetch = [&](int k0) {
    if constexpr (!ABF16) {
      const float* A = (const float*)Av;
#pragma unroll
      for (int i = 0; i < 4; i++)
        aregf[i] = *(const float4*)(A + (size_t)(m0 + ar8 + i * 32) * K + k0 + ac4);
    } else {
      const unsigned short* A = (const unsigned short*)Av;
#pragma unroll
      for (int i = 0; i < 2; i++)
        aregh[i] = *(const u16x8*)(A + (size_t)(m0 + hr + i * 64) * K + k0 + hc8);
    }
#pragma unroll
    for (int i = 0; i < 2; i++)
      bregh[i] = *(const u16x8*)(Bw + (size_t)(n0 + hr + i * 64) * K + k0 + hc8);
  };

  auto stage = [&](int buf) {
    unsigned short* as = As[buf];
    unsigned short* bs = Bs[buf];
    if constexpr (!ABF16) {
#pragma unroll
      for (int i = 0; i < 4; i++) {
        float4 f = aregf[i];
        *(uint2*)&as[(ar8 + i * 32) * LDT + ac4] =
            make_uint2(pack2bf(f.x, f.y), pack2bf(f.z, f.w));
      }
    } else {
#pragma unroll
      for (int i = 0; i < 2; i++)
        *(u16x8*)&as[(hr + i * 64) * LDT + hc8] = aregh[i];
    }
#pragma unroll
    for (int i = 0; i < 2; i++)
      *(u16x8*)&bs[(hr + i * 64) * LDT + hc8] = bregh[i];
  };

  fetch(0);
  stage(0);

  const int rsel = lane & 15;
  const int aoff = (lane >> 4) << 3;    // A: K0..7/K16..23 vs K8..15/K24..31
  const int boff = (lane >> 4) << 3;    // B: K0..7/K16..23 vs K8..15/K24..31
  const int nIter = K / BK;

  for (int it = 0; it < nIter; ++it) {
    __syncthreads();
    if (it + 1 < nIter) fetch((it + 1) * BK);
    const int cur = it & 1;
    const unsigned short* as = As[cur];
    const unsigned short* bs = Bs[cur];

    Frag af[4], bfr[2];
#pragma unroll
    for (int mi = 0; mi < 4; mi++) {
      int r = wm * 64 + mi * 16 + rsel;
      af[mi].h[0] = *(const u16x8*)&as[r * LDT + aoff];
      af[mi].h[1] = *(const u16x8*)&as[r * LDT + aoff + 16];
    }
#pragma unroll
    for (int ni = 0; ni < 2; ni++) {
      int r = wn * 32 + ni * 16 + rsel;
      bfr[ni].h[0] = *(const u16x8*)&bs[r * LDT + boff];
      bfr[ni].h[1] = *(const u16x8*)&bs[r * LDT + boff + 16];
    }
#pragma unroll
    for (int mi = 0; mi < 4; mi++)
#pragma unroll
      for (int ni = 0; ni < 2; ni++)
        acc[mi][ni] = __builtin_amdgcn_wmma_f32_16x16x32_bf16(
            false, af[mi].v, false, bfr[ni].v, (short)0, acc[mi][ni], false, false);

    if (it + 1 < nIter) stage(cur ^ 1);   // other buffer: no barrier needed here
  }

  const int rb = m0 + wm * 64 + ((lane >> 4) << 3);
  const int cb = n0 + wn * 32 + (lane & 15);
#pragma unroll
  for (int mi = 0; mi < 4; mi++) {
#pragma unroll
    for (int ni = 0; ni < 2; ni++) {
#pragma unroll
      for (int v = 0; v < 8; v++) {
        int rr = rb + mi * 16 + v;
        int cc = cb + ni * 16;
        float val = acc[mi][ni][v] + bias[cc];
        if constexpr (EPI == 0) {
          ((float*)out0)[(size_t)rr * N + cc] = val;
        } else if constexpr (EPI == 1) {
          if (cc < DIMD) {
            ((unsigned short*)out0)[(size_t)rr * DIMD + cc] = f2bf(val);
          } else {
            int d = cc - DIMD, bb = rr >> 12, s = rr & (SEQ - 1);
            ((unsigned short*)out1)[(((size_t)(bb * DIMD + d)) << 12) + s] = f2bf(val);
          }
        } else {
          size_t o = (size_t)rr * DIMD + cc;
          ((unsigned short*)out0)[o] = f2bf(val * bf2f(qbuf[o]));
        }
      }
    }
  }
}

// --------------------------- gates (N=3 matvec) ----------------------------
__global__ __launch_bounds__(256) void gates_kernel(
    const float* __restrict__ input, const float* __restrict__ W_in,
    const float* __restrict__ b_in, float* __restrict__ gbuf)
{
  __shared__ float sw[3][DIMD];
  const int tid = threadIdx.x;
  for (int idx = tid; idx < 3 * DIMD; idx += 256) {
    int j = idx >> 10, k = idx & (DIMD - 1);
    sw[j][k] = W_in[(size_t)k * 2051 + 2048 + j];
  }
  __syncthreads();
  const int lane = tid & 31, wave = tid >> 5;
  const int m = blockIdx.x * 8 + wave;
  const float* row = input + (size_t)m * DIMD;
  float a0 = 0.f, a1 = 0.f, a2 = 0.f;
  for (int k = lane; k < DIMD; k += 32) {
    float x = row[k];
    a0 += x * sw[0][k]; a1 += x * sw[1][k]; a2 += x * sw[2][k];
  }
#pragma unroll
  for (int off = 16; off > 0; off >>= 1) {
    a0 += __shfl_down(a0, off, 32);
    a1 += __shfl_down(a1, off, 32);
    a2 += __shfl_down(a2, off, 32);
  }
  if (lane == 0) {
    int bb = m >> 12, s = m & (SEQ - 1);
    gbuf[((size_t)bb * 3 + 0) * SEQ + s] = a0 + b_in[2048];
    gbuf[((size_t)bb * 3 + 1) * SEQ + s] = a1 + b_in[2049];
    gbuf[((size_t)bb * 3 + 2) * SEQ + s] = a2 + b_in[2050];
  }
}

// --------------- depthwise conv chain + gelu + modulation mix --------------
constexpr int DT = 16, ST = 256;
constexpr int CH   = ST + 32;   // ctx tile cols (14 left halo, 18 right)
constexpr int C0N  = ST + 17;   // c0 values needed for c1 halo
constexpr int C0LD = ST + 24;

template<bool DO_SUM>
__global__ __launch_bounds__(256) void conv_kernel(
    const unsigned short* __restrict__ ctx_bf,     // (B,D,S) bf16
    const float* __restrict__ w0, const float* __restrict__ b0,
    const float* __restrict__ w1, const float* __restrict__ b1,
    const float* __restrict__ gbuf,                // (B,3,S)
    const float* __restrict__ gsum_in,             // (B,D) c1 row sums
    float* __restrict__ gsum_out,
    unsigned short* __restrict__ ctxall)           // (B,S,D) bf16
{
  __shared__ float sctx[DT][CH];
  __shared__ float sc0[DT][C0LD];
  __shared__ float sw0[DT][14];
  __shared__ float sw1[DT][18];
  __shared__ float sb0[DT], sb1[DT], sred[DT];
  __shared__ float sg[3][ST];

  const int tid = threadIdx.x;
  const int b  = blockIdx.z;
  const int d0 = blockIdx.y * DT;
  const int s0 = blockIdx.x * ST;

  for (int idx = tid; idx < DT * 14; idx += 256)
    sw0[idx / 14][idx % 14] = w0[(size_t)(d0 + idx / 14) * 14 + idx % 14];
  for (int idx = tid; idx < DT * 18; idx += 256)
    sw1[idx / 18][idx % 18] = w1[(size_t)(d0 + idx / 18) * 18 + idx % 18];
  if (tid < DT) { sb0[tid] = b0[d0 + tid]; sb1[tid] = b1[d0 + tid]; sred[tid] = 0.f; }
  if constexpr (!DO_SUM) {
    for (int idx = tid; idx < 3 * ST; idx += 256) {
      int j = idx / ST, s = idx % ST;
      sg[j][s] = gbuf[((size_t)b * 3 + j) * SEQ + s0 + s];
    }
  }
  for (int idx = tid; idx < DT * CH; idx += 256) {
    int d = idx / CH, ss = idx % CH;
    int s = s0 - 14 + ss;
    float v = 0.f;
    if (s >= 0 && s < SEQ) v = bf2f(ctx_bf[(((size_t)(b * DIMD + d0 + d)) << 12) + s]);
    sctx[d][ss] = v;
  }
  __syncthreads();

  // c0 = gelu(conv14(ctx)) on [s0-8, s0+ST+9)
  for (int idx = tid; idx < DT * C0N; idx += 256) {
    int d = idx / C0N, j = idx % C0N;
    float a = sb0[d];
#pragma unroll
    for (int i = 0; i < 14; i++) a += sctx[d][j + i] * sw0[d][i];
    sc0[d][j] = gelu(a);
  }
  __syncthreads();

  if constexpr (DO_SUM) {
    const int d = tid >> 4;
    float part = 0.f;
    for (int j = tid & 15; j < ST; j += 16) {
      float a = sb1[d];
#pragma unroll
      for (int i = 0; i < 18; i++) a += sc0[d][j + i] * sw1[d][i];
      part += gelu(a);
    }
    atomicAdd(&sred[d], part);
    __syncthreads();
    if (tid < DT) atomicAdd(&gsum_out[(size_t)b * DIMD + d0 + tid], sred[tid]);
  } else {
    const int d = tid & 15;
    const float gs = gelu(gsum_in[(size_t)b * DIMD + d0 + d] * (1.0f / (float)SEQ));
    for (int j = tid >> 4; j < ST; j += 16) {
      float a = sb1[d];
#pragma unroll
      for (int i = 0; i < 18; i++) a += sc0[d][j + i] * sw1[d][i];
      float c1 = gelu(a);
      float val = sc0[d][j + 8] * sg[0][j] + c1 * sg[1][j] + gs * sg[2][j];
      ctxall[(((size_t)(b * SEQ + s0 + j)) << 10) + d0 + d] = f2bf(val);   // coalesced 32B/row
    }
  }
}

// --------------------------- weight prep kernels ---------------------------
__global__ __launch_bounds__(256) void cvt_transpose(
    const float* __restrict__ src, unsigned short* __restrict__ dst,
    int ld, int coff, int K)
{
  int k = blockIdx.x * 256 + threadIdx.x;
  int n = blockIdx.y;
  if (k < K) dst[(size_t)n * K + k] = f2bf(src[(size_t)k * ld + coff + n]);
}
__global__ __launch_bounds__(256) void cvt_plain(
    const float* __restrict__ src, unsigned short* __restrict__ dst, int n)
{
  int i = blockIdx.x * 256 + threadIdx.x;
  if (i < n) dst[i] = f2bf(src[i]);
}

// ---------------------------------------------------------------------------
extern "C" void kernel_launch(void* const* d_in, const int* in_sizes, int n_in,
                              void* d_out, int out_size, void* d_ws, size_t ws_size,
                              hipStream_t stream) {
  (void)in_sizes; (void)n_in; (void)out_size; (void)ws_size;
  const float* input = (const float*)d_in[0];
  const float* W_in  = (const float*)d_in[1];
  const float* b_in  = (const float*)d_in[2];
  const float* w0    = (const float*)d_in[3];
  const float* b0    = (const float*)d_in[4];
  const float* w1    = (const float*)d_in[5];
  const float* b1    = (const float*)d_in[6];
  const float* W_ctx = (const float*)d_in[7];
  const float* b_ctx = (const float*)d_in[8];
  const float* W_out = (const float*)d_in[9];
  const float* b_out = (const float*)d_in[10];
  float* out = (float*)d_out;

  char* ws = (char*)d_ws;
  size_t off = 0;
  auto alloc = [&](size_t bytes) -> void* {
    void* p = ws + off; off += (bytes + 255) & ~(size_t)255; return p;
  };
  const size_t BSD = (size_t)BATCH * SEQ * DIMD;
  unsigned short* Wt_in  = (unsigned short*)alloc((size_t)2048 * 1024 * 2);
  unsigned short* Wt_ctx = (unsigned short*)alloc((size_t)1024 * 1024 * 2);
  unsigned short* Wt_out = (unsigned short*)alloc((size_t)1024 * 1024 * 2);
  float* gbuf            = (float*)alloc((size_t)BATCH * 3 * SEQ * 4);
  float* gsum            = (float*)alloc((size_t)BATCH * DIMD * 4);
  unsigned short* qbuf   = (unsigned short*)alloc(BSD * 2);
  unsigned short* ctxbf  = (unsigned short*)alloc(BSD * 2);
  unsigned short* ctxall = (unsigned short*)alloc(BSD * 2);
  unsigned short* ybuf   = (unsigned short*)alloc(BSD * 2);

  // weight prep (bf16, N-major for straight LDS staging)
  cvt_transpose<<<dim3(4, 2048), 256, 0, stream>>>(W_in,  Wt_in,  2051, 0, 1024);
  cvt_transpose<<<dim3(4, 1024), 256, 0, stream>>>(W_out, Wt_out, 1024, 0, 1024);
  cvt_plain<<<dim3(4096), 256, 0, stream>>>(W_ctx, Wt_ctx, 1024 * 1024);   // 'oc' already (N,K)

  // GEMM1: proj -> query (BS,D) + ctx (B,D,S)
  gemm_wmma<false, 1><<<dim3(16, 256), 256, 0, stream>>>(
      input, Wt_in, b_in, nullptr, qbuf, ctxbf, BATCH * SEQ, 2048, DIMD);

  // gates (last 3 columns of W_in)
  gates_kernel<<<dim3(4096), 256, 0, stream>>>(input, W_in, b_in, gbuf);

  // conv chain: pass A accumulates c1 row sums, pass B emits context_all
  hipMemsetAsync(gsum, 0, (size_t)BATCH * DIMD * 4, stream);
  conv_kernel<true><<<dim3(16, 64, 8), 256, 0, stream>>>(
      ctxbf, w0, b0, w1, b1, gbuf, nullptr, gsum, nullptr);
  conv_kernel<false><<<dim3(16, 64, 8), 256, 0, stream>>>(
      ctxbf, w0, b0, w1, b1, gbuf, gsum, nullptr, ctxall);

  // GEMM3: modulator = ctxall @ W_ctx^T + b_ctx; y = query * modulator (bf16)
  gemm_wmma<true, 2><<<dim3(8, 256), 256, 0, stream>>>(
      ctxall, Wt_ctx, b_ctx, qbuf, ybuf, nullptr, BATCH * SEQ, DIMD, DIMD);

  // GEMM4: out = y @ W_out + b_out (f32)
  gemm_wmma<true, 0><<<dim3(8, 256), 256, 0, stream>>>(
      ybuf, Wt_out, b_out, nullptr, out, nullptr, BATCH * SEQ, DIMD, DIMD);
}
